// NCMOpenMax_65927747994107
// MI455X (gfx1250) — matrix operand order repaired and checked by reading, not compiled, not run on hardware
//
#include <hip/hip_runtime.h>
#include <hip/hip_bf16.h>

typedef __attribute__((ext_vector_type(16))) _Float16 v16h;
typedef __attribute__((ext_vector_type(8)))  _Float16 v8h;
typedef __attribute__((ext_vector_type(4)))  _Float16 v4h;
typedef __attribute__((ext_vector_type(8)))  float    v8f;
typedef __attribute__((ext_vector_type(4)))  float    v4f;

#define D     128
#define KCOLS 512
#define EPSN  1e-8f
#define LOG2E 1.4426950408889634074f

// ---------------------------------------------------------------------------
// Kernel 1: normalize muK rows (f32 -> f16, row-major into workspace).
// One wave per row; 4 floats per lane; full-wave shfl_xor reduction.
// ---------------------------------------------------------------------------
__global__ void muk_normalize_kernel(const float* __restrict__ muK,
                                     _Float16* __restrict__ mn) {
    const int k    = blockIdx.x;      // muK row (0..511)
    const int lane = threadIdx.x;     // 0..31
    const float* row = muK + (size_t)k * D;

    v4f a = *(const v4f*)(row + lane * 4);
    float ss = a[0]*a[0] + a[1]*a[1] + a[2]*a[2] + a[3]*a[3];
    #pragma unroll
    for (int m = 1; m <= 16; m <<= 1) ss += __shfl_xor(ss, m, 32);

    const float scale = 1.0f / fmaxf(sqrtf(ss), EPSN);

    v4h h;
    #pragma unroll
    for (int j = 0; j < 4; ++j) h[j] = (_Float16)(a[j] * scale);
    *(v4h*)(mn + (size_t)k * D + lane * 4) = h;
}

// ---------------------------------------------------------------------------
// Kernel 2: fused normalize(X) -> cosine scores (WMMA f16 -> f32 acc)
//           -> streaming softmax (scores bounded in [-1,1], no max pass).
// log2(e) is folded into X's normalization scale so the accumulator is
// score*log2e and exp() is a single v_exp_f32 (2^x).
// Block = 128 threads = 4 waves; wave handles 16 rows x 512 cols.
// Dynamic LDS: 64*512 f32 exp-scores + 64 f32 reciprocal row sums.
// ---------------------------------------------------------------------------
__global__ void cosine_softmax_kernel(const float* __restrict__ X,
                                      const _Float16* __restrict__ mn,
                                      float* __restrict__ out) {
    extern __shared__ float smem[];                 // [64*512] + [64]
    float* sinv = smem + 64 * KCOLS;

    const int tid    = threadIdx.x;
    const int lane   = tid & 31;
    const int wave   = tid >> 5;
    const int half16 = lane >> 4;                   // 0: lanes 0-15, 1: 16-31
    const int l16    = lane & 15;

    // ---- Load this wave's 16 X rows; lane L and L+16 split row (l16). ----
    // A-fragment (16-bit, 16x32): lanes 0-15 hold K {0..7,16..23},
    // lanes 16-31 hold K {8..15,24..31} (per k-tile of 32).
    const int rowBase = blockIdx.x * 64 + wave * 16;
    const float* xrow = X + (size_t)(rowBase + l16) * D;
    const int aoff = half16 * 8;

    float xbuf[64];
    #pragma unroll
    for (int t = 0; t < 4; ++t) {
        const int d0 = t * 32 + aoff;
        v4f q0 = __builtin_nontemporal_load((const v4f*)(xrow + d0));
        v4f q1 = __builtin_nontemporal_load((const v4f*)(xrow + d0 + 4));
        v4f q2 = __builtin_nontemporal_load((const v4f*)(xrow + d0 + 16));
        v4f q3 = __builtin_nontemporal_load((const v4f*)(xrow + d0 + 20));
        #pragma unroll
        for (int j = 0; j < 4; ++j) {
            xbuf[t*16 + j]      = q0[j];
            xbuf[t*16 + 4 + j]  = q1[j];
            xbuf[t*16 + 8 + j]  = q2[j];
            xbuf[t*16 + 12 + j] = q3[j];
        }
    }
    float ss = 0.0f;
    #pragma unroll
    for (int i = 0; i < 64; ++i) ss += xbuf[i] * xbuf[i];
    ss += __shfl_xor(ss, 16, 32);                   // combine the two row halves
    // Fold log2(e) into the scale: WMMA then yields score*log2(e).
    const float scale = LOG2E / fmaxf(sqrtf(ss), EPSN);

    v16h afrag[4];
    #pragma unroll
    for (int t = 0; t < 4; ++t) {
        #pragma unroll
        for (int e = 0; e < 16; ++e)
            afrag[t][e] = (_Float16)(xbuf[t*16 + e] * scale);
    }

    // ---- Loop over 32 column tiles (16 mu-rows each). ----
    // B-fragment (16-bit dense 32x16): lane = N, K matches A;
    // lanes 0-15 K {0..7,16..23}, lanes 16-31 K {8..15,24..31} per k-tile.
    const int boff = half16 * 8;
    float rs[8];
    #pragma unroll
    for (int i = 0; i < 8; ++i) rs[i] = 0.0f;

    float* swave = smem + wave * 16 * KCOLS;

    for (int c = 0; c < 32; ++c) {
        const _Float16* mrow = mn + (size_t)(c * 16 + l16) * D;
        v8f acc = {0.f, 0.f, 0.f, 0.f, 0.f, 0.f, 0.f, 0.f};
        #pragma unroll
        for (int t = 0; t < 4; ++t) {
            union { v16h v; v8h h[2]; } b;
            b.h[0] = *(const v8h*)(mrow + t * 32 + boff);
            b.h[1] = *(const v8h*)(mrow + t * 32 + boff + 16);
            acc = __builtin_amdgcn_wmma_f32_16x16x32_f16(
                false, afrag[t], false, b.v, (short)0, acc, false, false);
        }
        // C/D layout: VGPR i -> row (i + half16*8), col = lane%16 within tile.
        #pragma unroll
        for (int i = 0; i < 8; ++i) {
            const float e = __builtin_amdgcn_exp2f(acc[i]); // 2^(s*log2e)=exp(s)
            rs[i] += e;
            swave[(i + half16 * 8) * KCOLS + (c * 16 + l16)] = e;
        }
    }

    // ---- Row sums: reduce across the 16 lanes holding each row's N. ----
    #pragma unroll
    for (int m = 1; m <= 8; m <<= 1) {
        #pragma unroll
        for (int i = 0; i < 8; ++i) rs[i] += __shfl_xor(rs[i], m, 32);
    }
    if (l16 == 0) {
        #pragma unroll
        for (int i = 0; i < 8; ++i)
            sinv[wave * 16 + half16 * 8 + i] = 1.0f / rs[i];
    }
    __syncthreads();

    // ---- Coalesced vectorized writeout: out = exp(score) / rowsum. ----
    // 512 cols/row, float4 never straddles a row (512/4 = 128 quads/row).
    const v4f* s4 = (const v4f*)smem;
    v4f* o4 = (v4f*)(out + (size_t)blockIdx.x * 64 * KCOLS);
    #pragma unroll 4
    for (int e = tid; e < 64 * KCOLS / 4; e += 128) {
        v4f v = s4[e];
        v *= sinv[e >> 7];                          // row = (e*4) >> 9
        __builtin_nontemporal_store(v, o4 + e);     // write-once: bypass caches
    }
}

extern "C" void kernel_launch(void* const* d_in, const int* in_sizes, int n_in,
                              void* d_out, int out_size, void* d_ws, size_t ws_size,
                              hipStream_t stream) {
    const float* X   = (const float*)d_in[0];
    const float* muK = (const float*)d_in[1];
    float* out       = (float*)d_out;
    _Float16* mn     = (_Float16*)d_ws;             // 512*128 f16 = 128 KB

    const int N = in_sizes[0] / D;                  // 262144
    const int K = in_sizes[1] / D;                  // 512

    muk_normalize_kernel<<<K, 32, 0, stream>>>(muK, mn);

    const size_t smem_bytes = (size_t)(64 * KCOLS + 64) * sizeof(float); // ~128 KB
    hipFuncSetAttribute((const void*)cosine_softmax_kernel,
                        hipFuncAttributeMaxDynamicSharedMemorySize,
                        (int)smem_bytes);
    cosine_softmax_kernel<<<N / 64, 128, smem_bytes, stream>>>(X, mn, out);
}
